// MultiScaleDecoderAdapter_41601053229108
// MI455X (gfx1250) — matrix operand order repaired and checked by reading, not compiled, not run on hardware
//
#include <hip/hip_runtime.h>
#include <hip/hip_bf16.h>
#include <math.h>

// ---------- types ----------
typedef __bf16 bf16_t;
typedef bf16_t v16bf __attribute__((ext_vector_type(16)));
typedef float  v8f   __attribute__((ext_vector_type(8)));
typedef int    i4    __attribute__((ext_vector_type(4)));
typedef float  f4    __attribute__((ext_vector_type(4)));
// GCC-style vector type matching the async-LDS builtin's expected parameter
typedef int v4i_t __attribute__((vector_size(16)));
typedef __attribute__((address_space(1))) v4i_t* g_v4i_p;
typedef __attribute__((address_space(3))) v4i_t* l_v4i_p;

union BFrag { v16bf f; i4 q[2]; };

__device__ __forceinline__ unsigned short f2bf(float f) {
    unsigned int u = __float_as_uint(f);
    u += 0x7FFFu + ((u >> 16) & 1u);           // round-to-nearest-even
    return (unsigned short)(u >> 16);
}
__device__ __forceinline__ float bf2f(unsigned short u) {
    return __uint_as_float(((unsigned int)u) << 16);
}

#define B_  8
#define CH_ 512
#define H_  128
#define W_  128
#define HID 64
#define NPIX (H_*W_)            // 16384
#define NBPIX (B_*NPIX)         // 131072

#if __has_builtin(__builtin_amdgcn_global_load_async_to_lds_b128)
#define HAVE_ASYNC_LDS 1
#else
#define HAVE_ASYNC_LDS 0
#endif

// ---------- 1) bilinear resize cond 64x64 -> 128x128, NCHW f32 -> NHWC bf16 ----------
__global__ void k_resize_cond(const float* __restrict__ cond, unsigned short* __restrict__ condr) {
    int tid = blockIdx.x * blockDim.x + threadIdx.x;
    if (tid >= B_*H_*W_*128) return;
    int c = tid & 127;
    int x = (tid >> 7) & 127;
    int y = (tid >> 14) & 127;
    int b = tid >> 21;
    float sy = y * 0.5f - 0.25f;
    float sx = x * 0.5f - 0.25f;
    int y0f = (int)floorf(sy); float fy = sy - (float)y0f;
    int x0f = (int)floorf(sx); float fx = sx - (float)x0f;
    int y0 = min(max(y0f, 0), 63), y1 = min(max(y0f + 1, 0), 63);
    int x0 = min(max(x0f, 0), 63), x1 = min(max(x0f + 1, 0), 63);
    const float* src = cond + ((size_t)(b * 128 + c)) * 64 * 64;
    float v00 = src[y0*64+x0], v01 = src[y0*64+x1];
    float v10 = src[y1*64+x0], v11 = src[y1*64+x1];
    float v = (1.f-fy)*((1.f-fx)*v00 + fx*v01) + fy*((1.f-fx)*v10 + fx*v11);
    condr[tid] = f2bf(v);
}

// ---------- 2) pack OIHW f32 conv weights -> WMMA-A layout [O, 9*Cin] bf16, k = t*Cin + c ----------
__global__ void k_pack_w(const float* __restrict__ w, unsigned short* __restrict__ A, int Cin) {
    int tid = blockIdx.x * blockDim.x + threadIdx.x;
    int total = HID * Cin * 9;
    if (tid >= total) return;
    int c = tid % Cin;
    int t = (tid / Cin) % 9;
    int o = tid / (Cin * 9);
    int dy = t / 3, dx = t % 3;
    float v = w[((o * Cin + c) * 3 + dy) * 3 + dx];
    A[(size_t)o * (9 * Cin) + t * Cin + c] = f2bf(v);
}

// ---------- 3) implicit-GEMM 3x3 conv + bias + SiLU via WMMA, LDS-staged weights ----------
// one wave per (b, h, 16-pixel tile); 4 accumulator tiles cover 64 out channels.
// scalar (SGPR) control flow for the row bound so EXEC stays all-ones around WMMA.
template<int CIN>
__global__ __launch_bounds__(256) void k_conv3x3_wmma(const unsigned short* __restrict__ in,
                                                      const unsigned short* __restrict__ wA,
                                                      const float* __restrict__ bias,
                                                      unsigned short* __restrict__ out) {
    constexpr int KTOT = 9 * CIN;
    __shared__ unsigned short sA[HID * KTOT];   // conv1: 147456 B, conv2: 73728 B (<=320KB LDS)

    // --- stage packed weights into LDS (async global->LDS when available) ---
    const int nElem = HID * KTOT;
#if HAVE_ASYNC_LDS
    for (int i = threadIdx.x * 8; i < nElem; i += 256 * 8) {
        __builtin_amdgcn_global_load_async_to_lds_b128(
            (g_v4i_p)(wA + i), (l_v4i_p)(&sA[i]), 0, 0);
    }
#if __has_builtin(__builtin_amdgcn_s_wait_asynccnt)
    __builtin_amdgcn_s_wait_asynccnt(0);
#else
    asm volatile("s_wait_asynccnt 0" ::: "memory");
#endif
#else
    for (int i = threadIdx.x * 8; i < nElem; i += 256 * 8)
        *(i4*)(&sA[i]) = *(const i4*)(wA + i);
#endif
    __syncthreads();

    int wave = blockIdx.x * 8 + (threadIdx.x >> 5);
    int lane = threadIdx.x & 31;
    // wave-uniform coordinates -> force into SGPRs for scalar branching
    int w0S = __builtin_amdgcn_readfirstlane((wave & 7) * 16);
    int hS  = __builtin_amdgcn_readfirstlane((wave >> 3) & 127);
    int bS  = __builtin_amdgcn_readfirstlane(wave >> 10);
    int n    = lane & 15;
    int half = lane >> 4;

    v8f acc[4];
#pragma unroll
    for (int ot = 0; ot < 4; ++ot)
#pragma unroll
        for (int r = 0; r < 8; ++r) acc[ot][r] = 0.f;

    for (int dy = -1; dy <= 1; ++dy) {
        int y = hS + dy;
        if (y < 0 || y > (H_ - 1)) continue;      // scalar s_cbranch, EXEC untouched
        for (int dx = -1; dx <= 1; ++dx) {
            int t = (dy + 1) * 3 + (dx + 1);
            int x = w0S + n + dx;
            bool xok = (unsigned)x < (unsigned)W_;
#pragma unroll
            for (int c0 = 0; c0 < CIN; c0 += 32) {
                BFrag Bf;
                if (xok) {
                    const unsigned short* p =
                        in + (((size_t)(bS * H_ + y) * W_ + x) * CIN + c0 + half * 8);
                    Bf.q[0] = *(const i4*)(p);
                    Bf.q[1] = *(const i4*)(p + 16);
                } else {
                    Bf.q[0] = (i4)0;
                    Bf.q[1] = (i4)0;
                }
                int klo = t * CIN + c0 + half * 8;
#pragma unroll
                for (int ot = 0; ot < 4; ++ot) {
                    const unsigned short* ap = &sA[(ot * 16 + n) * KTOT + klo];
                    BFrag Af;
                    Af.q[0] = *(const i4*)(ap);        // K .. K+7       (ds_load_b128)
                    Af.q[1] = *(const i4*)(ap + 16);   // K+16 .. K+23
                    acc[ot] = __builtin_amdgcn_wmma_f32_16x16x32_bf16(
                        false, Af.f, false, Bf.f, (short)0, acc[ot], false, false);
                }
            }
        }
    }
    // epilogue: bias + SiLU, store NHWC bf16
    size_t obase = ((size_t)(bS * H_ + hS) * W_ + (w0S + n)) * HID;
#pragma unroll
    for (int ot = 0; ot < 4; ++ot)
#pragma unroll
        for (int r = 0; r < 8; ++r) {
            int oc = ot * 16 + half * 8 + r;
            float v = acc[ot][r] + bias[oc];
            v = v / (1.f + __expf(-v));
            out[obase + oc] = f2bf(v);
        }
}

// ---------- 4) per-pixel 1x1 heads on cf: alpha/beta/edge ----------
__global__ void k_maps(const unsigned short* __restrict__ cf,
                       const float* __restrict__ wA, const float* __restrict__ bA,
                       const float* __restrict__ wB, const float* __restrict__ bB,
                       const float* __restrict__ wE, const float* __restrict__ bE,
                       float* __restrict__ alpha, float* __restrict__ beta,
                       float* __restrict__ ew) {
    int p = blockIdx.x * blockDim.x + threadIdx.x;
    if (p >= NBPIX) return;
    const unsigned short* v = cf + (size_t)p * HID;
    float sa = 0.f, sb = 0.f, se = 0.f;
    for (int c = 0; c < HID; ++c) {
        float x = bf2f(v[c]);
        sa += x * wA[c]; sb += x * wB[c]; se += x * wE[c];
    }
    alpha[p] = 0.35f * tanhf(sa + bA[0]);
    beta[p]  = 0.10f * tanhf(sb + bB[0]);
    ew[p]    = 1.f / (1.f + __expf(-(se + bE[0])));
}

// ---------- 5) channel mean/std of hidden (coalesced over w, prefetch ahead) ----------
__global__ void k_stats(const float* __restrict__ hidden,
                        float* __restrict__ hm, float* __restrict__ hs) {
    int bh = blockIdx.x;              // b*128 + h
    int w  = threadIdx.x;             // 128 threads
    int b  = bh >> 7, h = bh & 127;
    const float* base = hidden + (size_t)b * CH_ * NPIX + (size_t)h * W_ + w;
    float s = 0.f, s2 = 0.f;
    for (int c = 0; c < CH_; ++c) {
        __builtin_prefetch(base + (size_t)(c + 16) * NPIX, 0, 1);   // global_prefetch_b8
        float x = base[(size_t)c * NPIX];
        s += x; s2 += x * x;
    }
    float m = s * (1.f / CH_);
    float var = s2 * (1.f / CH_) - m * m;
    if (var < 0.f) var = 0.f;
    hm[bh * W_ + w] = m;
    hs[bh * W_ + w] = sqrtf(var);
}

// ---------- 6) 8x8 adaptive pools ----------
__global__ void k_pool_cf(const unsigned short* __restrict__ cf, float* __restrict__ cs) {
    int tid = blockIdx.x * blockDim.x + threadIdx.x;
    if (tid >= B_ * 256 * HID) return;
    int c  = tid & 63;
    int px = (tid >> 6) & 15;
    int py = (tid >> 10) & 15;
    int b  = tid >> 14;
    float s = 0.f;
    for (int i = 0; i < 8; ++i)
        for (int j = 0; j < 8; ++j)
            s += bf2f(cf[((size_t)(b * H_ + py * 8 + i) * W_ + px * 8 + j) * HID + c]);
    cs[tid] = s * (1.f / 64.f);       // layout [b, py*16+px, c], c fastest
}

__global__ void k_pool_stats(const float* __restrict__ hm, const float* __restrict__ hs,
                             float* __restrict__ hsml) {
    int tid = blockIdx.x * blockDim.x + threadIdx.x;
    if (tid >= B_ * 2 * 256) return;
    int j = tid & 255;
    int s = (tid >> 8) & 1;
    int b = tid >> 9;
    int py = j >> 4, px = j & 15;
    const float* src = (s ? hs : hm) + (size_t)b * NPIX;
    float acc = 0.f;
    for (int i = 0; i < 8; ++i)
        for (int jj = 0; jj < 8; ++jj)
            acc += src[(py * 8 + i) * W_ + px * 8 + jj];
    hsml[tid] = acc * (1.f / 64.f);   // layout [b, stat, 256]
}

// ---------- 7) tiny 4-head cross-attention, LDS-staged K/V ----------
__global__ void k_attn(const float* __restrict__ cs, const float* __restrict__ hsml,
                       const float* __restrict__ w_q, const float* __restrict__ b_q,
                       const float* __restrict__ w_k, const float* __restrict__ b_k,
                       const float* __restrict__ w_v, const float* __restrict__ b_v,
                       float* __restrict__ ctxc) {
    __shared__ float kx[8][256];
    __shared__ float vx[8][256];
    int bh = blockIdx.x;              // b*4 + head
    int b = bh >> 2, head = bh & 3;
    int j = threadIdx.x;              // 256 threads: one per position
    const float* csj = cs + ((size_t)b * 256 + j) * HID;
#pragma unroll
    for (int d = 0; d < 8; ++d) {
        int oc = head * 8 + d;
        float sk = b_k[oc], sv = b_v[oc];
        for (int c = 0; c < HID; ++c) {
            float x = csj[c];
            sk += w_k[oc * HID + c] * x;
            sv += w_v[oc * HID + c] * x;
        }
        kx[d][j] = sk; vx[d][j] = sv;
    }
    float q[8];
    float h0 = hsml[(b * 2 + 0) * 256 + j];
    float h1 = hsml[(b * 2 + 1) * 256 + j];
#pragma unroll
    for (int d = 0; d < 8; ++d) {
        int oc = head * 8 + d;
        q[d] = w_q[oc * 2 + 0] * h0 + w_q[oc * 2 + 1] * h1 + b_q[oc];
    }
    __syncthreads();
    const float scale = 1.f / (sqrtf(8.f) + 1e-6f);
    float mx = -3.4e38f;
    for (int jj = 0; jj < 256; ++jj) {
        float l = 0.f;
#pragma unroll
        for (int d = 0; d < 8; ++d) l += q[d] * kx[d][jj];
        mx = fmaxf(mx, l * scale);
    }
    float se = 0.f;
    float cacc[8];
#pragma unroll
    for (int d = 0; d < 8; ++d) cacc[d] = 0.f;
    for (int jj = 0; jj < 256; ++jj) {
        float l = 0.f;
#pragma unroll
        for (int d = 0; d < 8; ++d) l += q[d] * kx[d][jj];
        float e = __expf(l * scale - mx);
        se += e;
#pragma unroll
        for (int d = 0; d < 8; ++d) cacc[d] += e * vx[d][jj];
    }
    float inv = 1.f / se;
#pragma unroll
    for (int d = 0; d < 8; ++d)
        ctxc[((size_t)(b * 32) + head * 8 + d) * 256 + j] = cacc[d] * inv;
}

// ---------- 8) fold w_out over 32 ctx channels (commutes with bilinear resize) ----------
__global__ void k_ctx_combine(const float* __restrict__ ctxc, const float* __restrict__ w_out,
                              const float* __restrict__ b_out, float* __restrict__ ctxw) {
    int tid = blockIdx.x * blockDim.x + threadIdx.x;
    if (tid >= B_ * 256) return;
    int j = tid & 255, b = tid >> 8;
    float s = b_out[0];
    for (int d = 0; d < 32; ++d)
        s += w_out[d] * ctxc[((size_t)b * 32 + d) * 256 + j];
    ctxw[tid] = s;
}

// ---------- 9) upsample ctx 16->128 and build hm2 = (1+a)(hm + g*ctx) + beta ----------
__global__ void k_hm2map(const float* __restrict__ ctxw,
                         const float* __restrict__ alpha, const float* __restrict__ beta,
                         const float* __restrict__ hm,
                         const float* __restrict__ gamma, const int* __restrict__ lidx,
                         float* __restrict__ ctxmap, float* __restrict__ hm2) {
    int p = blockIdx.x * blockDim.x + threadIdx.x;
    if (p >= NBPIX) return;
    int b = p >> 14;
    int pp = p & 16383;
    int y = pp >> 7, x = pp & 127;
    float sy = y * 0.125f - 0.4375f;
    float sx = x * 0.125f - 0.4375f;
    int y0f = (int)floorf(sy); float fy = sy - (float)y0f;
    int x0f = (int)floorf(sx); float fx = sx - (float)x0f;
    int y0 = min(max(y0f, 0), 15), y1 = min(max(y0f + 1, 0), 15);
    int x0 = min(max(x0f, 0), 15), x1 = min(max(x0f + 1, 0), 15);
    const float* cw = ctxw + b * 256;
    float v = (1.f-fy)*((1.f-fx)*cw[y0*16+x0] + fx*cw[y0*16+x1])
            +      fy *((1.f-fx)*cw[y1*16+x0] + fx*cw[y1*16+x1]);
    ctxmap[p] = v;
    float g = gamma[lidx[0]];
    hm2[p] = (1.f + alpha[p]) * (hm[p] + g * v) + beta[p];
}

// ---------- 10) per-pixel coefficients: out = hidden*A + B ----------
__global__ void k_coeffs(const float* __restrict__ hm2, const float* __restrict__ ctxmap,
                         const float* __restrict__ alpha, const float* __restrict__ beta,
                         const float* __restrict__ ew,
                         const float* __restrict__ gamma, const int* __restrict__ lidx,
                         float* __restrict__ Ac, float* __restrict__ Bc) {
    int p = blockIdx.x * blockDim.x + threadIdx.x;
    if (p >= NBPIX) return;
    int b = p >> 14;
    int pp = p & 16383;
    int y = pp >> 7, x = pp & 127;
    const float* src = hm2 + (size_t)b * NPIX;
    float s = 0.f;
    for (int dy = -1; dy <= 1; ++dy)
        for (int dx = -1; dx <= 1; ++dx) {
            int yy = y + dy, xx = x + dx;
            if ((unsigned)yy < (unsigned)H_ && (unsigned)xx < (unsigned)W_)
                s += src[yy * W_ + xx];
        }
    float hp = hm2[p] - s * (1.f / 9.f);
    float g = gamma[lidx[0]];
    float a1 = 1.f + alpha[p];
    Ac[p] = a1;
    Bc[p] = g * ctxmap[p] * a1 + beta[p] + ew[p] * hp;
}

// ---------- 11) final memory-bound streaming pass, float4, non-temporal ----------
__global__ void k_final(const float* __restrict__ hidden,
                        const float* __restrict__ Ac, const float* __restrict__ Bc,
                        float* __restrict__ out) {
    size_t idx = (size_t)blockIdx.x * blockDim.x + threadIdx.x;
    size_t total4 = (size_t)B_ * CH_ * NPIX / 4;
    if (idx >= total4) return;
    size_t i4_ = idx * 4;
    int b = (int)(i4_ >> 23);                 // 512*16384 = 2^23 per batch
    int p = (int)(i4_ & 16383);
    size_t pix = ((size_t)b << 14) + p;       // 4-aligned, no plane crossing
    f4 hv = __builtin_nontemporal_load((const f4*)(hidden + i4_));
    f4 Av = *(const f4*)(Ac + pix);
    f4 Bv = *(const f4*)(Bc + pix);
    f4 o;
#pragma unroll
    for (int i = 0; i < 4; ++i) o[i] = hv[i] * Av[i] + Bv[i];
    __builtin_nontemporal_store(o, (f4*)(out + i4_));
}

// ---------- launch ----------
extern "C" void kernel_launch(void* const* d_in, const int* in_sizes, int n_in,
                              void* d_out, int out_size, void* d_ws, size_t ws_size,
                              hipStream_t stream) {
    const float* hidden  = (const float*)d_in[0];
    const float* cond    = (const float*)d_in[1];
    const float* w_pre1  = (const float*)d_in[2];
    const float* b_pre1  = (const float*)d_in[3];
    const float* w_pre2  = (const float*)d_in[4];
    const float* b_pre2  = (const float*)d_in[5];
    const float* w_alpha = (const float*)d_in[6];
    const float* b_alpha = (const float*)d_in[7];
    const float* w_beta  = (const float*)d_in[8];
    const float* b_beta  = (const float*)d_in[9];
    const float* w_edge  = (const float*)d_in[10];
    const float* b_edge  = (const float*)d_in[11];
    const float* w_q     = (const float*)d_in[12];
    const float* b_q     = (const float*)d_in[13];
    const float* w_k     = (const float*)d_in[14];
    const float* b_k     = (const float*)d_in[15];
    const float* w_v     = (const float*)d_in[16];
    const float* b_v     = (const float*)d_in[17];
    const float* w_out   = (const float*)d_in[18];
    const float* b_out   = (const float*)d_in[19];
    const float* gamma   = (const float*)d_in[20];
    const int*   lidx    = (const int*)d_in[21];
    float* out = (float*)d_out;

    size_t off = 0;
    auto alloc = [&](size_t bytes) -> void* {
        void* p = (char*)d_ws + off;
        off += (bytes + 255) & ~(size_t)255;
        return p;
    };
    unsigned short* condr = (unsigned short*)alloc((size_t)B_*H_*W_*128 * 2);  // 33.5 MB
    unsigned short* A1    = (unsigned short*)alloc((size_t)HID*1152 * 2);
    unsigned short* A2    = (unsigned short*)alloc((size_t)HID*576 * 2);
    unsigned short* cf1   = (unsigned short*)alloc((size_t)B_*H_*W_*HID * 2);  // 16.8 MB
    unsigned short* cf2   = (unsigned short*)alloc((size_t)B_*H_*W_*HID * 2);  // 16.8 MB
    float* alpha  = (float*)alloc((size_t)NBPIX * 4);
    float* beta   = (float*)alloc((size_t)NBPIX * 4);
    float* ew     = (float*)alloc((size_t)NBPIX * 4);
    float* hm     = (float*)alloc((size_t)NBPIX * 4);
    float* hs     = (float*)alloc((size_t)NBPIX * 4);
    float* cs     = (float*)alloc((size_t)B_*256*HID * 4);
    float* hsml   = (float*)alloc((size_t)B_*2*256 * 4);
    float* ctxc   = (float*)alloc((size_t)B_*32*256 * 4);
    float* ctxw   = (float*)alloc((size_t)B_*256 * 4);
    float* ctxmap = (float*)alloc((size_t)NBPIX * 4);
    float* hm2    = (float*)alloc((size_t)NBPIX * 4);
    float* Ac     = (float*)alloc((size_t)NBPIX * 4);
    float* Bc     = (float*)alloc((size_t)NBPIX * 4);

    // cond branch (bf16 WMMA path)
    k_resize_cond<<<(B_*H_*W_*128 + 255) / 256, 256, 0, stream>>>(cond, condr);
    k_pack_w<<<(HID*128*9 + 255) / 256, 256, 0, stream>>>(w_pre1, A1, 128);
    k_pack_w<<<(HID*64*9 + 255) / 256, 256, 0, stream>>>(w_pre2, A2, 64);
    // 8192 waves, 8 waves/block -> 1024 blocks
    k_conv3x3_wmma<128><<<1024, 256, 0, stream>>>(condr, A1, b_pre1, cf1);
    k_conv3x3_wmma<64><<<1024, 256, 0, stream>>>(cf1, A2, b_pre2, cf2);
    k_maps<<<(NBPIX + 255) / 256, 256, 0, stream>>>(cf2, w_alpha, b_alpha, w_beta, b_beta,
                                                    w_edge, b_edge, alpha, beta, ew);
    // hidden stats
    k_stats<<<B_*H_, 128, 0, stream>>>(hidden, hm, hs);
    // pools + attention
    k_pool_cf<<<(B_*256*HID + 255) / 256, 256, 0, stream>>>(cf2, cs);
    k_pool_stats<<<(B_*2*256 + 255) / 256, 256, 0, stream>>>(hm, hs, hsml);
    k_attn<<<B_*4, 256, 0, stream>>>(cs, hsml, w_q, b_q, w_k, b_k, w_v, b_v, ctxc);
    k_ctx_combine<<<(B_*256 + 255) / 256, 256, 0, stream>>>(ctxc, w_out, b_out, ctxw);
    // per-pixel coefficient folding
    k_hm2map<<<(NBPIX + 255) / 256, 256, 0, stream>>>(ctxw, alpha, beta, hm, gamma, lidx,
                                                      ctxmap, hm2);
    k_coeffs<<<(NBPIX + 255) / 256, 256, 0, stream>>>(hm2, ctxmap, alpha, beta, ew,
                                                      gamma, lidx, Ac, Bc);
    // final memory-bound pass
    size_t total4 = (size_t)B_ * CH_ * NPIX / 4;
    k_final<<<(unsigned)((total4 + 255) / 256), 256, 0, stream>>>(hidden, Ac, Bc, out);
    (void)in_sizes; (void)n_in; (void)out_size; (void)ws_size;
}
